// RealLidarTokenizer_27075473834735
// MI455X (gfx1250) — compile-verified, matrix operand order checked by reading, not run on hardware
//
#include <hip/hip_runtime.h>
#include <hip/hip_bf16.h>

// ---------------- problem constants (match reference) ----------------
#define BB 4
#define CC 5
#define HH 64
#define DD 256
#define GX 128
#define GY 128
#define GZ 1
#define VV (GX * GY * GZ)   // 16384
#define GV (VV + 1)         // +1 dump row for invalid points (same trick as reference)
#define EPSV 1e-5f
#define PTS_PER_BLOCK 128   // 8 waves -> 8 M-tiles of 16

typedef _Float16 v8h  __attribute__((ext_vector_type(8)));
typedef _Float16 v16h __attribute__((ext_vector_type(16)));
typedef float    v8f  __attribute__((ext_vector_type(8)));

static __device__ __forceinline__ v16h pack16(v8h lo, v8h hi) {
    v16h r;
#pragma unroll
    for (int i = 0; i < 8; ++i) { r[i] = lo[i]; r[i + 8] = hi[i]; }
    return r;
}

// =====================================================================
// Kernel A: fused MLP (5->64 VALU, 64->256 WMMA f16) + voxel atomic-max
// =====================================================================
__global__ __launch_bounds__(256) void fused_mlp_voxmax(
    const float* __restrict__ points,
    const float* __restrict__ W1, const float* __restrict__ b1,
    const float* __restrict__ g1, const float* __restrict__ be1,
    const float* __restrict__ m1, const float* __restrict__ v1,
    const float* __restrict__ W2, const float* __restrict__ b2,
    const float* __restrict__ g2, const float* __restrict__ be2,
    const float* __restrict__ m2, const float* __restrict__ v2,
    unsigned int* __restrict__ grid_max, int* __restrict__ cnt, int N)
{
    // LDS staging
    __shared__ __align__(16) _Float16 W2t[DD * HH];              // 32 KB, W2^T [n][k]
    __shared__ __align__(16) _Float16 Atile[PTS_PER_BLOCK * HH]; // 16 KB, layer1 out
    __shared__ float W1s[CC * HH];
    __shared__ float sc1[HH], sh1[HH];
    __shared__ float sc2[DD], sh2[DD];
    __shared__ int   segs[PTS_PER_BLOCK];

    const int t  = threadIdx.x;
    const int b  = blockIdx.y;
    const int m0 = blockIdx.x * PTS_PER_BLOCK;

    // ---- phase 0: stage weights / folded BN params ----
    for (int i = t; i < DD * HH; i += 256) {           // coalesced read of W2[k][n]
        int k = i >> 8, n = i & (DD - 1);
        W2t[n * HH + k] = (_Float16)W2[i];
    }
    for (int i = t; i < CC * HH; i += 256) W1s[i] = W1[i];
    if (t < HH) {
        float s = g1[t] * rsqrtf(v1[t] + EPSV);
        sc1[t] = s;
        sh1[t] = (b1[t] - m1[t]) * s + be1[t];
    }
    if (t < DD) {
        float s = g2[t] * rsqrtf(v2[t] + EPSV);
        sc2[t] = s;
        sh2[t] = (b2[t] - m2[t]) * s + be2[t];
    }
    __syncthreads();

    // ---- phase 1: layer 1 (K=5) + voxelization ----
    {
        const int pl   = t & (PTS_PER_BLOCK - 1);
        const int half = t >> 7;               // 0 or 1 -> which 32 h-dims
        const int p    = m0 + pl;
        float x[CC] = {0.f, 0.f, 0.f, 0.f, 0.f};
        if (p < N) {
            const float* pp = points + ((size_t)b * N + p) * CC;
#pragma unroll
            for (int c = 0; c < CC; ++c) x[c] = pp[c];
        }
        if (half == 0) {
            int s = VV;                        // dump row (invalid / out-of-range)
            if (p < N) {
                bool valid = (x[0] >= -51.2f) & (x[0] < 51.2f) &
                             (x[1] >= -51.2f) & (x[1] < 51.2f) &
                             (x[2] >= -5.0f)  & (x[2] < 3.0f);
                if (valid) {
                    int cx = (int)floorf((x[0] + 51.2f) * 1.25f);
                    int cy = (int)floorf((x[1] + 51.2f) * 1.25f);
                    int cz = (int)floorf((x[2] + 5.0f) * 0.125f);
                    s = cx * (GY * GZ) + cy * GZ + cz;
                }
                atomicAdd(&cnt[b * GV + s], 1);   // dump row absorbs invalid
            }
            segs[pl] = s;
        }
#pragma unroll
        for (int jj = 0; jj < 32; ++jj) {
            int j = half * 32 + jj;
            float acc = x[0] * W1s[0 * HH + j] + x[1] * W1s[1 * HH + j] +
                        x[2] * W1s[2 * HH + j] + x[3] * W1s[3 * HH + j] +
                        x[4] * W1s[4 * HH + j];
            float y = fmaxf(acc * sc1[j] + sh1[j], 0.0f);
            Atile[pl * HH + j] = (_Float16)y;
        }
    }
    __syncthreads();

    // ---- phase 2: 64->256 via v_wmma_f32_16x16x32_f16 ----
    const int w    = t >> 5;          // wave id 0..7 -> M-tile
    const int lane = t & 31;
    const int r    = lane & 15;       // A row within tile / B,C,D column
    const int hh   = lane >> 4;       // lane-half -> K(/M) grouping

    // A fragments per ISA 16-bit A layout: VGPR0-3 = K base+half*8, VGPR4-7 = +16
    const _Float16* Ar = &Atile[(w * 16 + r) * HH];
    v16h a0 = pack16(*(const v8h*)(Ar +  0 + hh * 8), *(const v8h*)(Ar + 16 + hh * 8));
    v16h a1 = pack16(*(const v8h*)(Ar + 32 + hh * 8), *(const v8h*)(Ar + 48 + hh * 8));

    // Precompute the 8 destination row bases (C/D VGPR i -> M = i + hh*8).
    // seg == VV targets the dump row, so the atomics below are UNCONDITIONAL:
    // no per-lane exec-mask branching in the epilogue.
    unsigned int* rowp[8];
#pragma unroll
    for (int i = 0; i < 8; ++i) {
        int s = segs[w * 16 + hh * 8 + i];
        rowp[i] = grid_max + ((size_t)b * GV + s) * DD;
    }

#pragma unroll 2
    for (int nt = 0; nt < DD / 16; ++nt) {
        const _Float16* Br = &W2t[(nt * 16 + r) * HH];   // B: lane = column n
        v16h bf0 = pack16(*(const v8h*)(Br +  0 + hh * 8), *(const v8h*)(Br + 16 + hh * 8));
        v16h bf1 = pack16(*(const v8h*)(Br + 32 + hh * 8), *(const v8h*)(Br + 48 + hh * 8));
        v8f c = {};
        c = __builtin_amdgcn_wmma_f32_16x16x32_f16(false, a0, false, bf0,
                                                   (short)0, c, false, false);
        c = __builtin_amdgcn_wmma_f32_16x16x32_f16(false, a1, false, bf1,
                                                   (short)0, c, false, false);
        const int d  = nt * 16 + r;
        const float sc = sc2[d], sh = sh2[d];
#pragma unroll
        for (int i = 0; i < 8; ++i) {
            float y = fmaxf(c[i] * sc + sh, 0.0f);   // relu >= 0 -> u32-orderable
            atomicMax(rowp[i] + d, __float_as_uint(y));
        }
    }
}

// =====================================================================
// Kernel B: per-sample occupancy prefix scan -> pos, num
// =====================================================================
__global__ __launch_bounds__(256) void occ_scan(
    const int* __restrict__ cnt, int* __restrict__ pos, int* __restrict__ num)
{
    __shared__ int s[256];
    const int b = blockIdx.x, t = threadIdx.x;
    const int* c = cnt + (size_t)b * GV;     // dump row (index VV) ignored
    const int base = t * (VV / 256);
    int local = 0;
#pragma unroll 4
    for (int i = 0; i < VV / 256; ++i) local += (c[base + i] > 0) ? 1 : 0;
    s[t] = local;
    __syncthreads();
    for (int off = 1; off < 256; off <<= 1) {   // Hillis-Steele inclusive scan
        int v = (t >= off) ? s[t - off] : 0;
        __syncthreads();
        s[t] += v;
        __syncthreads();
    }
    int run = s[t] - local;                      // exclusive prefix
    if (t == 255) num[b] = s[255];
    int* pp = pos + (size_t)b * VV;
    for (int i = 0; i < VV / 256; ++i) {
        if (c[base + i] > 0) pp[base + i] = run++;
    }
}

// =====================================================================
// Kernel C: compact scatter (tok, crd) + padding mask
// =====================================================================
__global__ __launch_bounds__(256) void compact_scatter(
    const unsigned int* __restrict__ grid_max, const int* __restrict__ cnt,
    const int* __restrict__ pos, const int* __restrict__ num,
    float* __restrict__ tok, float* __restrict__ crd, float* __restrict__ mask)
{
    const int v = blockIdx.x, b = blockIdx.y, t = threadIdx.x;
    const size_t bv  = (size_t)b * VV + v;   // output/pos index
    const size_t bvg = (size_t)b * GV + v;   // grid/cnt index (stride V+1)
    if (t == 0) mask[bv] = (v >= num[b]) ? 1.0f : 0.0f;
    if (cnt[bvg] > 0) {
        const int p = pos[bv];
        const size_t dst = (size_t)b * VV + p;
        tok[dst * DD + t] = __uint_as_float(grid_max[bvg * DD + t]);
        if (t < 3) {
            const float vs[3] = {0.8f, 0.8f, 8.0f};
            const float lo[3] = {-51.2f, -51.2f, -5.0f};
            const int ci = (t == 0) ? (v / (GY * GZ)) : (t == 1) ? ((v / GZ) % GY) : (v % GZ);
            crd[dst * 3 + t] = (float)ci * vs[t] + 0.5f * vs[t] + lo[t];
        }
    }
}

// =====================================================================
extern "C" void kernel_launch(void* const* d_in, const int* in_sizes, int n_in,
                              void* d_out, int out_size, void* d_ws, size_t ws_size,
                              hipStream_t stream) {
    const float* points = (const float*)d_in[0];
    const float* W1  = (const float*)d_in[1];
    const float* b1  = (const float*)d_in[2];
    const float* g1  = (const float*)d_in[3];
    const float* be1 = (const float*)d_in[4];
    const float* m1  = (const float*)d_in[5];
    const float* v1  = (const float*)d_in[6];
    const float* W2  = (const float*)d_in[7];
    const float* b2  = (const float*)d_in[8];
    const float* g2  = (const float*)d_in[9];
    const float* be2 = (const float*)d_in[10];
    const float* m2  = (const float*)d_in[11];
    const float* v2  = (const float*)d_in[12];

    const int N = in_sizes[0] / (BB * CC);   // 120000

    // workspace layout: grid_max (B*(V+1)*D u32) | cnt (B*(V+1)) | pos (B*V) | num (B)
    const size_t gridElems = (size_t)BB * GV * DD;
    unsigned int* grid_max = (unsigned int*)d_ws;
    int* cnt = (int*)(grid_max + gridElems);
    int* pos = cnt + (size_t)BB * GV;
    int* num = pos + (size_t)BB * VV;

    // output layout: tok (B,V,D) | crd (B,V,3) | mask (B,V), all f32
    float* tok  = (float*)d_out;
    float* crd  = tok + (size_t)BB * VV * DD;
    float* mask = crd + (size_t)BB * VV * 3;

    hipMemsetAsync(d_ws, 0, (gridElems + (size_t)BB * GV) * sizeof(unsigned int), stream);
    hipMemsetAsync(d_out, 0, (size_t)out_size * sizeof(float), stream);

    dim3 gA((N + PTS_PER_BLOCK - 1) / PTS_PER_BLOCK, BB);
    fused_mlp_voxmax<<<gA, 256, 0, stream>>>(points, W1, b1, g1, be1, m1, v1,
                                             W2, b2, g2, be2, m2, v2,
                                             grid_max, cnt, N);
    occ_scan<<<BB, 256, 0, stream>>>(cnt, pos, num);
    dim3 gC(VV, BB);
    compact_scatter<<<gC, 256, 0, stream>>>(grid_max, cnt, pos, num, tok, crd, mask);
}
